// MultiHeadAttention_64467459113184
// MI455X (gfx1250) — matrix operand order repaired and checked by reading, not compiled, not run on hardware
//
#include <hip/hip_runtime.h>
#include <hip/hip_bf16.h>
#include <math.h>

typedef __attribute__((ext_vector_type(16))) __bf16 v16bf;
typedef __attribute__((ext_vector_type(8)))  __bf16 v8bf;
typedef __attribute__((ext_vector_type(4)))  __bf16 v4bf;
typedef __attribute__((ext_vector_type(8)))  float  v8f;
typedef __attribute__((ext_vector_type(4)))  int    v4i;

#define D_MODEL 1024
#define NUM_HEADS 16
#define D_K 64
#define BATCH 2
#define SEQ 2048
#define M_TOTAL (BATCH * SEQ)   // 4096

union U16 { v16bf v; v8bf h2[2]; __bf16 e[16]; };

// ---------------------------------------------------------------------------
// CDNA5 async global->LDS copy (ASYNCcnt path), with synchronous fallback
// ---------------------------------------------------------------------------
#if defined(__has_builtin)
#if __has_builtin(__builtin_amdgcn_global_load_async_to_lds_b128) && \
    __has_builtin(__builtin_amdgcn_s_wait_asynccnt)
#define USE_ASYNC_LDS 1
#endif
#endif
#ifndef USE_ASYNC_LDS
#define USE_ASYNC_LDS 0
#endif

__device__ __forceinline__ void async_copy16(const __bf16* __restrict__ g, __bf16* l) {
#if USE_ASYNC_LDS
  __builtin_amdgcn_global_load_async_to_lds_b128(
      (__attribute__((address_space(1))) v4i*)g,
      (__attribute__((address_space(3))) v4i*)l, 0, 0);
#else
  *(v8bf*)l = *(const v8bf*)g;
#endif
}
__device__ __forceinline__ void async_wait() {
#if USE_ASYNC_LDS
  __builtin_amdgcn_s_wait_asynccnt(0);
#endif
}

// ---------------------------------------------------------------------------
// fp32 -> bf16 conversion (vectorized x4)
// ---------------------------------------------------------------------------
__global__ __launch_bounds__(256) void cvt_f32_bf16(const float* __restrict__ s,
                                                    __bf16* __restrict__ d, int n4) {
  int i = blockIdx.x * blockDim.x + threadIdx.x;
  if (i < n4) {
    float4 f = ((const float4*)s)[i];
    v4bf o;
    o[0] = (__bf16)f.x; o[1] = (__bf16)f.y; o[2] = (__bf16)f.z; o[3] = (__bf16)f.w;
    ((v4bf*)d)[i] = o;
  }
}

// ---------------------------------------------------------------------------
// GEMM: C[4096,1024] = A[4096,1024] @ W[1024,1024]^T + bias
// Block 256 threads = 8 waves (4M x 2N); block tile 128x128, wave tile 32x64
// (2x4 WMMA tiles, 8 wmma per 32-wide k-step). A and W k-slices are staged
// into double-buffered LDS with async global->LDS copies.
// MODE 0: bf16 row-major  MODE 1: bf16 Vt[B,H,dk,S]  MODE 2: fp32 row-major
// ---------------------------------------------------------------------------
template <int MODE>
__global__ __launch_bounds__(256) void gemm_wmma_bf16(
    const __bf16* __restrict__ A, const __bf16* __restrict__ W,
    const float* __restrict__ bias, void* __restrict__ out) {
  __shared__ __align__(16) __bf16 As[2][128][40];   // 32 data cols + pad
  __shared__ __align__(16) __bf16 Ws[2][128][40];

  const int tid  = threadIdx.x;
  const int lane = tid & 31;
  const int l15  = lane & 15;
  const int hi   = lane >> 4;
  const int wid  = tid >> 5;
  const int mblk = blockIdx.y * 128, nblk = blockIdx.x * 128;
  const int moff = (wid >> 1) * 32;    // wave M offset in block
  const int noff = (wid & 1) * 64;     // wave N offset in block

  v8f acc[2][4] = {};

  // stage one 128x32 k-slice of A and of W: 512 16B chunks each, 2+2 per thread
  auto stage = [&](int buf, int kk) {
#pragma unroll
    for (int i = 0; i < 2; ++i) {
      int c = tid + i * 256;
      int row = c >> 2, col = (c & 3) * 8;
      async_copy16(A + (size_t)(mblk + row) * D_MODEL + kk + col, &As[buf][row][col]);
      async_copy16(W + (size_t)(nblk + row) * D_MODEL + kk + col, &Ws[buf][row][col]);
    }
  };

  stage(0, 0);
  for (int kk = 0; kk < D_MODEL; kk += 32) {
    const int buf = (kk >> 5) & 1;
    async_wait();
    __syncthreads();                      // staged data visible to all waves
    if (kk + 32 < D_MODEL) stage(buf ^ 1, kk + 32);

    U16 af[2], bfr[4];
#pragma unroll
    for (int mt = 0; mt < 2; ++mt) {
      const __bf16* ap = &As[buf][moff + mt * 16 + l15][0];
      af[mt].h2[0] = *(const v8bf*)(ap + 8 * hi);
      af[mt].h2[1] = *(const v8bf*)(ap + 16 + 8 * hi);
    }
#pragma unroll
    for (int nt = 0; nt < 4; ++nt) {
      const __bf16* wp = &Ws[buf][noff + nt * 16 + l15][0];
      bfr[nt].h2[0] = *(const v8bf*)(wp + 8 * hi);
      bfr[nt].h2[1] = *(const v8bf*)(wp + 16 + 8 * hi);
    }
#pragma unroll
    for (int mt = 0; mt < 2; ++mt)
#pragma unroll
      for (int nt = 0; nt < 4; ++nt)
        acc[mt][nt] = __builtin_amdgcn_wmma_f32_16x16x32_bf16(
            false, af[mt].v, false, bfr[nt].v, (short)0, acc[mt][nt], false, false);
    __syncthreads();                      // all reads done before next overwrite
  }

  float bv[4];
#pragma unroll
  for (int nt = 0; nt < 4; ++nt) bv[nt] = bias[nblk + noff + nt * 16 + l15];

#pragma unroll
  for (int mt = 0; mt < 2; ++mt)
#pragma unroll
    for (int nt = 0; nt < 4; ++nt)
#pragma unroll
      for (int r = 0; r < 8; ++r) {
        float v = acc[mt][nt][r] + bv[nt];
        int m = mblk + moff + mt * 16 + 8 * hi + r;   // D row = r + 8*hi
        int n = nblk + noff + nt * 16 + l15;          // D col = lane&15
        if (MODE == 0) {
          ((__bf16*)out)[(size_t)m * D_MODEL + n] = (__bf16)v;
        } else if (MODE == 1) {
          int b = m >> 11, s = m & (SEQ - 1);
          int h = n >> 6,  d = n & (D_K - 1);
          ((__bf16*)out)[((size_t)((b * NUM_HEADS + h) * D_K + d)) * SEQ + s] = (__bf16)v;
        } else {
          ((float*)out)[(size_t)m * D_MODEL + n] = v;
        }
      }
}

// ---------------------------------------------------------------------------
// Flash attention: 4 waves/block share one (b,h); each wave owns a 16-query
// tile. Per 32-key chunk, K (32x64) and V^T (64x32) tiles are staged once
// into double-buffered LDS (async) and consumed by all 4 waves.
// S^T = K.Q^T (keys=rows, query=lane) maps the exp'd score fragment into the
// B operand of ctx^T = V^T.P^T using only each lane's own values.
// ---------------------------------------------------------------------------
__global__ __launch_bounds__(128) void attn_flash_wmma(
    const __bf16* __restrict__ Q, const __bf16* __restrict__ K,
    const __bf16* __restrict__ Vt, __bf16* __restrict__ Ctx) {
  __shared__ __align__(16) __bf16 Ks[2][32][72];   // 64 data cols + pad
  __shared__ __align__(16) __bf16 Vs[2][64][40];   // 32 data cols + pad

  const int tid  = threadIdx.x;
  const int lane = tid & 31;
  const int l15  = lane & 15;
  const int hi   = lane >> 4;
  const int wvid = tid >> 5;
  const int bh   = blockIdx.x >> 5;                 // 0..31  (B*H)
  const int h = bh & (NUM_HEADS - 1), b = bh >> 4;
  const int qtile = (blockIdx.x & 31) * 4 + wvid;   // 0..127
  const int qbase = qtile * 16;
  const float scale = 0.125f;                       // 1/sqrt(64)

  // stage K tile (32 rows x 64) and V^T tile (64 rows x 32): 2+2 chunks/thread
  auto stage = [&](int buf, int j0) {
#pragma unroll
    for (int i = 0; i < 2; ++i) {
      int c = tid + i * 128;
      int kr = c >> 3, kc = (c & 7) * 8;            // K: 32 rows, 8 chunks/row
      async_copy16(K + (size_t)(b * SEQ + j0 + kr) * D_MODEL + h * D_K + kc,
                   &Ks[buf][kr][kc]);
      int vr = c >> 2, vc = (c & 3) * 8;            // V: 64 rows, 4 chunks/row
      async_copy16(Vt + (size_t)(bh * D_K + vr) * SEQ + j0 + vc,
                   &Vs[buf][vr][vc]);
    }
  };

  // Q as B-operand fragments (persistent)
  U16 qf[2];
  const __bf16* qrow = Q + (size_t)(b * SEQ + qbase + l15) * D_MODEL + h * D_K;
  qf[0].h2[0] = *(const v8bf*)(qrow + 8 * hi);
  qf[0].h2[1] = *(const v8bf*)(qrow + 16 + 8 * hi);
  qf[1].h2[0] = *(const v8bf*)(qrow + 32 + 8 * hi);
  qf[1].h2[1] = *(const v8bf*)(qrow + 48 + 8 * hi);

  v8f acc[4] = {};
  float mi = -3.0e38f, li = 0.0f;

  stage(0, 0);
  for (int j0 = 0; j0 < SEQ; j0 += 32) {
    const int buf = (j0 >> 5) & 1;
    async_wait();
    __syncthreads();
    if (j0 + 32 < SEQ) stage(buf ^ 1, j0 + 32);

    v8f sc[2] = {};
#pragma unroll
    for (int kb = 0; kb < 2; ++kb) {
      const __bf16* kr = &Ks[buf][kb * 16 + l15][0];
      U16 a0, a1;
      a0.h2[0] = *(const v8bf*)(kr + 8 * hi);
      a0.h2[1] = *(const v8bf*)(kr + 16 + 8 * hi);
      a1.h2[0] = *(const v8bf*)(kr + 32 + 8 * hi);
      a1.h2[1] = *(const v8bf*)(kr + 48 + 8 * hi);
      sc[kb] = __builtin_amdgcn_wmma_f32_16x16x32_bf16(
          false, a0.v, false, qf[0].v, (short)0, sc[kb], false, false);
      sc[kb] = __builtin_amdgcn_wmma_f32_16x16x32_bf16(
          false, a1.v, false, qf[1].v, (short)0, sc[kb], false, false);
    }

    // online softmax over this 32-key chunk (per query = per lane)
    float cmax = -3.0e38f;
#pragma unroll
    for (int r = 0; r < 8; ++r) {
      sc[0][r] *= scale; sc[1][r] *= scale;
      cmax = fmaxf(cmax, fmaxf(sc[0][r], sc[1][r]));
    }
    cmax = fmaxf(cmax, __shfl_xor(cmax, 16));
    float mnew = fmaxf(mi, cmax);
    float corr = __expf(mi - mnew);
    float p0[8], p1[8], psum = 0.0f;
#pragma unroll
    for (int r = 0; r < 8; ++r) {
      p0[r] = __expf(sc[0][r] - mnew);
      p1[r] = __expf(sc[1][r] - mnew);
      psum += p0[r] + p1[r];
    }
    psum += __shfl_xor(psum, 16);
    li = li * corr + psum;
    mi = mnew;
#pragma unroll
    for (int t = 0; t < 4; ++t)
#pragma unroll
      for (int r = 0; r < 8; ++r) acc[t][r] *= corr;

    // pack P^T into B operand (each lane's own p0/p1; o0/o1 unused)
    U16 pb;
#pragma unroll
    for (int j = 0; j < 8; ++j) {
      float o0 = __shfl_xor(p0[j], 16);
      float o1 = __shfl_xor(p1[j], 16);
      pb.e[j]     = (__bf16)p0[j];
      pb.e[8 + j] = (__bf16)p1[j];
    }

    // ctx^T += V^T . P^T from LDS-resident V tile
#pragma unroll
    for (int t = 0; t < 4; ++t) {
      const __bf16* vr = &Vs[buf][t * 16 + l15][0];
      U16 va;
      va.h2[0] = *(const v8bf*)(vr + 8 * hi);
      va.h2[1] = *(const v8bf*)(vr + 16 + 8 * hi);
      acc[t] = __builtin_amdgcn_wmma_f32_16x16x32_bf16(
          false, va.v, false, pb.v, (short)0, acc[t], false, false);
    }
    __syncthreads();
  }

  float inv = 1.0f / li;
#pragma unroll
  for (int t = 0; t < 4; ++t) {
    v8bf o;
#pragma unroll
    for (int r = 0; r < 8; ++r) o[r] = (__bf16)(acc[t][r] * inv);
    __bf16* cp = Ctx + (size_t)(b * SEQ + qbase + l15) * D_MODEL + h * D_K + t * 16 + 8 * hi;
    *(v8bf*)cp = o;
  }
}

// ---------------------------------------------------------------------------
// launch
// ---------------------------------------------------------------------------
extern "C" void kernel_launch(void* const* d_in, const int* in_sizes, int n_in,
                              void* d_out, int out_size, void* d_ws, size_t ws_size,
                              hipStream_t stream) {
  const float* Qin = (const float*)d_in[0];
  const float* Kin = (const float*)d_in[1];
  const float* Vin = (const float*)d_in[2];
  const float* Wq  = (const float*)d_in[3];
  const float* bq  = (const float*)d_in[4];
  const float* Wk  = (const float*)d_in[5];
  const float* bk  = (const float*)d_in[6];
  const float* Wv  = (const float*)d_in[7];
  const float* bv  = (const float*)d_in[8];
  const float* Wo  = (const float*)d_in[9];
  const float* bo  = (const float*)d_in[10];

  const size_t ACT = (size_t)M_TOTAL * D_MODEL;   // 4 Mi elems
  const size_t WEL = (size_t)D_MODEL * D_MODEL;   // 1 Mi elems
  char* ws = (char*)d_ws;
  __bf16* Xq  = (__bf16*)(ws);
  __bf16* Xk  = (__bf16*)(ws + 2 * ACT);
  __bf16* Xv  = (__bf16*)(ws + 4 * ACT);
  __bf16* Wqb = (__bf16*)(ws + 6 * ACT);
  __bf16* Wkb = (__bf16*)(ws + 6 * ACT + 2 * WEL);
  __bf16* Wvb = (__bf16*)(ws + 6 * ACT + 4 * WEL);
  __bf16* Wob = (__bf16*)(ws + 6 * ACT + 6 * WEL);
  __bf16* Qb  = (__bf16*)(ws + 6 * ACT + 8 * WEL);
  __bf16* Kb  = (__bf16*)(ws + 8 * ACT + 8 * WEL);
  __bf16* Vtb = (__bf16*)(ws + 10 * ACT + 8 * WEL);
  __bf16* Ctx = (__bf16*)(ws + 12 * ACT + 8 * WEL);   // 64 MiB total

  const int a4 = (int)(ACT / 4), w4 = (int)(WEL / 4);
  cvt_f32_bf16<<<a4 / 256, 256, 0, stream>>>(Qin, Xq, a4);
  cvt_f32_bf16<<<a4 / 256, 256, 0, stream>>>(Kin, Xk, a4);
  cvt_f32_bf16<<<a4 / 256, 256, 0, stream>>>(Vin, Xv, a4);
  cvt_f32_bf16<<<w4 / 256, 256, 0, stream>>>(Wq, Wqb, w4);
  cvt_f32_bf16<<<w4 / 256, 256, 0, stream>>>(Wk, Wkb, w4);
  cvt_f32_bf16<<<w4 / 256, 256, 0, stream>>>(Wv, Wvb, w4);
  cvt_f32_bf16<<<w4 / 256, 256, 0, stream>>>(Wo, Wob, w4);

  dim3 ggrid(D_MODEL / 128, M_TOTAL / 128);
  gemm_wmma_bf16<0><<<ggrid, 256, 0, stream>>>(Xq, Wqb, bq, Qb);
  gemm_wmma_bf16<0><<<ggrid, 256, 0, stream>>>(Xk, Wkb, bk, Kb);
  gemm_wmma_bf16<1><<<ggrid, 256, 0, stream>>>(Xv, Wvb, bv, Vtb);

  attn_flash_wmma<<<1024, 128, 0, stream>>>(Qb, Kb, Vtb, Ctx);

  gemm_wmma_bf16<2><<<ggrid, 256, 0, stream>>>(Ctx, Wob, bo, (float*)d_out);
}
